// RoPEAttention2D_1623497638214
// MI455X (gfx1250) — matrix operand order repaired and checked by reading, not compiled, not run on hardware
//
#include <hip/hip_runtime.h>
#include <hip/hip_bf16.h>

#define NSEQ 1024
#define CIN  768
#define NH   12
#define HD   64

typedef __attribute__((ext_vector_type(16))) _Float16 v16h;
typedef __attribute__((ext_vector_type(8)))  _Float16 v8h;
typedef __attribute__((ext_vector_type(8)))  float    v8f;

static __device__ __forceinline__ v16h mk16(v8h lo, v8h hi) {
  return __builtin_shufflevector(lo, hi, 0,1,2,3,4,5,6,7,8,9,10,11,12,13,14,15);
}
static __device__ __forceinline__ v8f zero8() {
  v8f z;
#pragma unroll
  for (int i = 0; i < 8; ++i) z[i] = 0.0f;
  return z;
}
static __device__ __forceinline__ v8f wmma16(v16h a, v16h b, v8f c) {
  // D = A(16x32) * B(32x16) + C, f32 accumulate
  return __builtin_amdgcn_wmma_f32_16x16x32_f16(false, a, false, b, (short)0, c, false, false);
}
// A-fragment: lane (m = lane&15, kh = lane>>4) holds contiguous K chunk [kh*16, kh*16+16)
static __device__ __forceinline__ v16h ldA_f16(const _Float16* p) {
  return mk16(*(const v8h*)p, *(const v8h*)(p + 8));
}
// B-fragment: lane (n = lane&15, kh = lane>>4) holds contiguous K chunk [kh*16, kh*16+16)
static __device__ __forceinline__ v16h ldB_f16(const _Float16* p) {
  return *(const v16h*)p;  // 32B aligned at all call sites
}

// ---------------------------------------------------------------------------
// Kernel 0: one-shot f32 -> f16 conversion (x, qkv_w, proj_w). Removes all
// per-K-step cvt VALU from the GEMM inner loops and halves L2 A/B traffic.
// ---------------------------------------------------------------------------
__global__ __launch_bounds__(256)
void cvt_f16_kernel(const float* __restrict__ src, _Float16* __restrict__ dst, int n) {
  const int i = (blockIdx.x * 256 + threadIdx.x) * 8;
  if (i < n) {
    v8f f = *(const v8f*)(src + i);
    *(v8h*)(dst + i) = __builtin_convertvector(f, v8h);
  }
}

// ---------------------------------------------------------------------------
// Kernel 1: QKV GEMM (f16 WMMA, f32 acc) + fused 2D-RoPE + LayerNorm epilogue.
// Tile: 128 rows x 64 cols (one head-slot of q/k/v). grid = (36, 128).
// q,k -> [B][12][1024][64] f16 (q pre-scaled by d^-0.5). v -> transposed [B][12][64][1024].
// ---------------------------------------------------------------------------
__global__ __launch_bounds__(256)
void qkv_rope_ln_kernel(const _Float16* __restrict__ xh, const _Float16* __restrict__ wh,
                        const float* __restrict__ qg, const float* __restrict__ qb,
                        const float* __restrict__ kg, const float* __restrict__ kb,
                        _Float16* __restrict__ qo, _Float16* __restrict__ ko,
                        _Float16* __restrict__ vt)
{
  __shared__ float tile[128 * 66];
  const int tid = threadIdx.x, lane = tid & 31, wid = tid >> 5;
  const int wm = wid & 3, wn = wid >> 2;         // 4 x 2 wave grid: 32 rows x 32 cols per wave
  const int lm = lane & 15, kh = lane >> 4, rh = lane >> 4;
  const int slot = blockIdx.x;                   // 0..35 = (which*12 + head)
  const int type = slot / NH, head = slot % NH;  // 0=q 1=k 2=v
  const int rowbase = blockIdx.y * 128;          // row in [0, 16384)

  v8f c[2][2];
#pragma unroll
  for (int i = 0; i < 2; ++i)
#pragma unroll
    for (int j = 0; j < 2; ++j) c[i][j] = zero8();

  for (int k0 = 0; k0 < CIN; k0 += 32) {
    v16h a[2], bfr[2];
#pragma unroll
    for (int s = 0; s < 2; ++s) {
      a[s]   = ldA_f16(xh + (size_t)(rowbase + wm * 32 + s * 16 + lm) * CIN + k0 + kh * 16);
      bfr[s] = ldB_f16(wh + (size_t)(slot * 64 + wn * 32 + s * 16 + lm) * CIN + k0 + kh * 16);
    }
    __builtin_prefetch(xh + (size_t)(rowbase + wm * 32 + lm) * CIN + k0 + 128, 0, 1);
#pragma unroll
    for (int i = 0; i < 2; ++i)
#pragma unroll
      for (int j = 0; j < 2; ++j) c[i][j] = wmma16(a[i], bfr[j], c[i][j]);
  }

  // C/D layout: lane col = lane&15, rows (lane>>4)*8 + v
#pragma unroll
  for (int i = 0; i < 2; ++i)
#pragma unroll
    for (int j = 0; j < 2; ++j)
#pragma unroll
      for (int v = 0; v < 8; ++v)
        tile[(wm * 32 + i * 16 + rh * 8 + v) * 66 + wn * 32 + j * 16 + lm] = c[i][j][v];
  __syncthreads();

  const int b_img = rowbase >> 10;
  const int nbase = rowbase & 1023;

  if (type == 2) {
    // store V transposed: vt[b][head][d][n]
    for (int idx = tid; idx < 128 * 64; idx += 256) {
      const int nn = idx & 127, d = idx >> 7;
      vt[((size_t)(b_img * NH + head) * HD + d) * NSEQ + nbase + nn] = (_Float16)tile[nn * 66 + d];
    }
  } else {
    if (tid < 128) {
      const int r = tid;
      const int n = nbase + r;
      const int hp = n >> 5, wp = n & 31;        // 2D position in 32x32 grid
      float vals[64];
      for (int d = 0; d < 64; ++d) vals[d] = tile[r * 66 + d];
      // interleaved-pair RoPE: dims [0,32) with pos=h, dims [32,64) with pos=w
#pragma unroll
      for (int i = 0; i < 16; ++i) {
        const float fr = __powf(10000.0f, -(float)i * (1.0f / 16.0f));
        float sh_, ch, sw, cw;
        __sincosf((float)hp * fr, &sh_, &ch);
        __sincosf((float)wp * fr, &sw, &cw);
        const float x0 = vals[2 * i], x1 = vals[2 * i + 1];
        vals[2 * i]     = x0 * ch - x1 * sh_;
        vals[2 * i + 1] = x1 * ch + x0 * sh_;
        const float y0 = vals[32 + 2 * i], y1 = vals[32 + 2 * i + 1];
        vals[32 + 2 * i]     = y0 * cw - y1 * sw;
        vals[32 + 2 * i + 1] = y1 * cw + y0 * sw;
      }
      // LayerNorm over head_dim
      float mu = 0.f;
      for (int d = 0; d < 64; ++d) mu += vals[d];
      mu *= (1.0f / 64.0f);
      float var = 0.f;
      for (int d = 0; d < 64; ++d) { const float t = vals[d] - mu; var += t * t; }
      var *= (1.0f / 64.0f);
      const float inv = rsqrtf(var + 1e-5f);
      const float* g  = (type == 0) ? qg : kg;
      const float* bi = (type == 0) ? qb : kb;
      const float sc  = (type == 0) ? 0.125f : 1.0f;  // fold HEAD_DIM^-0.5 into q
      for (int d = 0; d < 64; ++d)
        tile[r * 66 + d] = ((vals[d] - mu) * inv * g[d] + bi[d]) * sc;
    }
    __syncthreads();
    _Float16* dst = (type == 0) ? qo : ko;
    for (int idx = tid; idx < 128 * 64; idx += 256) {
      const int nn = idx >> 6, d = idx & 63;
      dst[((size_t)(b_img * NH + head) * NSEQ + nbase + nn) * HD + d] = (_Float16)tile[nn * 66 + d];
    }
  }
}

// ---------------------------------------------------------------------------
// Kernel 2: flash attention. grid = (8 q-blocks, 192 b*heads), 256 threads.
// Q[128x64] in LDS; per K/V block of 128: S via WMMA, online softmax in LDS,
// O += P*V via WMMA (V read transposed).
// ---------------------------------------------------------------------------
__global__ __launch_bounds__(256)
void flash_attn_kernel(const _Float16* __restrict__ qv, const _Float16* __restrict__ kv,
                       const _Float16* __restrict__ vt, _Float16* __restrict__ ao)
{
  __shared__ _Float16 qs[128 * 72];                           // 18 KB
  __shared__ __align__(16) unsigned char praw[128 * 136 * 2]; // 34 KB: S/P (f16), reused for O (f32)
  __shared__ float rowm[128], rowl[128], rowc[128];
  _Float16* sh = (_Float16*)praw;
  float* ob = (float*)praw;

  const int tid = threadIdx.x, lane = tid & 31, wid = tid >> 5;
  const int wm = wid & 3, wn = wid >> 2;
  const int lm = lane & 15, kh = lane >> 4, rh = lane >> 4;
  const int qblk = blockIdx.x, bh = blockIdx.y;

  const _Float16* qp = qv + ((size_t)bh * NSEQ + qblk * 128) * HD;
  const _Float16* kp = kv + (size_t)bh * NSEQ * HD;
  const _Float16* vp = vt + (size_t)bh * HD * NSEQ;

  for (int idx = tid; idx < 128 * 8; idx += 256) {
    const int nn = idx >> 3, c8 = idx & 7;
    *(v8h*)(qs + nn * 72 + c8 * 8) = *(const v8h*)(qp + nn * HD + c8 * 8);
  }
  if (tid < 128) { rowm[tid] = -1e30f; rowl[tid] = 0.0f; rowc[tid] = 0.0f; }

  v8f o[2][2];
#pragma unroll
  for (int i = 0; i < 2; ++i)
#pragma unroll
    for (int j = 0; j < 2; ++j) o[i][j] = zero8();
  __syncthreads();

  for (int kb = 0; kb < 8; ++kb) {
    // ---- S = Q * Kblk^T : per wave 32 rows x 64 cols ----
    v8f s[2][4];
#pragma unroll
    for (int i = 0; i < 2; ++i)
#pragma unroll
      for (int j = 0; j < 4; ++j) s[i][j] = zero8();
#pragma unroll
    for (int kk = 0; kk < 64; kk += 32) {
      v16h a[2];
#pragma unroll
      for (int si = 0; si < 2; ++si)
        a[si] = ldA_f16(qs + (wm * 32 + si * 16 + lm) * 72 + kk + kh * 16);
      v16h bf[4];
#pragma unroll
      for (int sj = 0; sj < 4; ++sj)
        bf[sj] = ldB_f16(kp + (size_t)(kb * 128 + wn * 64 + sj * 16 + lm) * HD + kk + kh * 16);
#pragma unroll
      for (int si = 0; si < 2; ++si)
#pragma unroll
        for (int sj = 0; sj < 4; ++sj) s[si][sj] = wmma16(a[si], bf[sj], s[si][sj]);
    }
#pragma unroll
    for (int si = 0; si < 2; ++si)
#pragma unroll
      for (int sj = 0; sj < 4; ++sj)
#pragma unroll
        for (int v = 0; v < 8; ++v)
          sh[(wm * 32 + si * 16 + rh * 8 + v) * 136 + wn * 64 + sj * 16 + lm] =
              (_Float16)s[si][sj][v];
    __syncthreads();

    // ---- online softmax row pass ----
    if (tid < 128) {
      _Float16* row = sh + tid * 136;
      float mx = -1e30f;
      for (int j = 0; j < 128; ++j) mx = fmaxf(mx, (float)row[j]);
      const float mnew = fmaxf(rowm[tid], mx);
      const float corr = __expf(rowm[tid] - mnew);
      float ssum = 0.f;
      for (int j = 0; j < 128; ++j) {
        const float e = __expf((float)row[j] - mnew);
        row[j] = (_Float16)e;
        ssum += e;
      }
      rowl[tid] = rowl[tid] * corr + ssum;
      rowm[tid] = mnew;
      rowc[tid] = corr;
    }
    __syncthreads();

    // ---- rescale O, then O += P * Vblk (K = 128) ----
#pragma unroll
    for (int si = 0; si < 2; ++si) {
      float cf[8];
#pragma unroll
      for (int v = 0; v < 8; ++v) cf[v] = rowc[wm * 32 + si * 16 + rh * 8 + v];
#pragma unroll
      for (int sj = 0; sj < 2; ++sj)
#pragma unroll
        for (int v = 0; v < 8; ++v) o[si][sj][v] *= cf[v];
    }
#pragma unroll
    for (int kk = 0; kk < 128; kk += 32) {
      v16h a[2];
#pragma unroll
      for (int si = 0; si < 2; ++si)
        a[si] = ldA_f16(sh + (wm * 32 + si * 16 + lm) * 136 + kk + kh * 16);
      v16h bf[2];
#pragma unroll
      for (int sj = 0; sj < 2; ++sj)
        bf[sj] = ldB_f16(vp + (size_t)(wn * 32 + sj * 16 + lm) * NSEQ + kb * 128 + kk + kh * 16);
#pragma unroll
      for (int si = 0; si < 2; ++si)
#pragma unroll
        for (int sj = 0; sj < 2; ++sj) o[si][sj] = wmma16(a[si], bf[sj], o[si][sj]);
    }
    __syncthreads();  // sh is rewritten next iteration / reused as ob below
  }

  // ---- normalize by 1/l, stage O (f32) into reused LDS, coalesced f16 store ----
#pragma unroll
  for (int si = 0; si < 2; ++si) {
    float lf[8];
#pragma unroll
    for (int v = 0; v < 8; ++v) lf[v] = 1.0f / rowl[wm * 32 + si * 16 + rh * 8 + v];
#pragma unroll
    for (int sj = 0; sj < 2; ++sj)
#pragma unroll
      for (int v = 0; v < 8; ++v)
        ob[(wm * 32 + si * 16 + rh * 8 + v) * 66 + wn * 32 + sj * 16 + lm] = o[si][sj][v] * lf[v];
  }
  __syncthreads();

  const int b_img = bh / NH, head = bh % NH;
  for (int idx = tid; idx < 128 * 64; idx += 256) {
    const int nn = idx >> 6, d = idx & 63;
    ao[((size_t)b_img * NSEQ + qblk * 128 + nn) * CIN + head * HD + d] = (_Float16)ob[nn * 66 + d];
  }
}

// ---------------------------------------------------------------------------
// Kernel 3: output projection GEMM. grid = (12, 128), tile 128x64, f32 out.
// ---------------------------------------------------------------------------
__global__ __launch_bounds__(256)
void proj_kernel(const _Float16* __restrict__ att, const _Float16* __restrict__ wh,
                 float* __restrict__ out)
{
  __shared__ float tile[128 * 66];
  const int tid = threadIdx.x, lane = tid & 31, wid = tid >> 5;
  const int wm = wid & 3, wn = wid >> 2;
  const int lm = lane & 15, kh = lane >> 4, rh = lane >> 4;
  const int colbase = blockIdx.x * 64;
  const int rowbase = blockIdx.y * 128;

  v8f c[2][2];
#pragma unroll
  for (int i = 0; i < 2; ++i)
#pragma unroll
    for (int j = 0; j < 2; ++j) c[i][j] = zero8();

  for (int k0 = 0; k0 < CIN; k0 += 32) {
    v16h a[2], bfr[2];
#pragma unroll
    for (int s = 0; s < 2; ++s) {
      a[s]   = ldA_f16(att + (size_t)(rowbase + wm * 32 + s * 16 + lm) * CIN + k0 + kh * 16);
      bfr[s] = ldB_f16(wh + (size_t)(colbase + wn * 32 + s * 16 + lm) * CIN + k0 + kh * 16);
    }
    __builtin_prefetch(att + (size_t)(rowbase + wm * 32 + lm) * CIN + k0 + 128, 0, 1);
#pragma unroll
    for (int i = 0; i < 2; ++i)
#pragma unroll
      for (int j = 0; j < 2; ++j) c[i][j] = wmma16(a[i], bfr[j], c[i][j]);
  }

#pragma unroll
  for (int i = 0; i < 2; ++i)
#pragma unroll
    for (int j = 0; j < 2; ++j)
#pragma unroll
      for (int v = 0; v < 8; ++v)
        tile[(wm * 32 + i * 16 + rh * 8 + v) * 66 + wn * 32 + j * 16 + lm] = c[i][j][v];
  __syncthreads();

  for (int idx = tid; idx < 128 * 64; idx += 256) {
    const int nn = idx >> 6, d = idx & 63;
    out[(size_t)(rowbase + nn) * CIN + colbase + d] = tile[nn * 66 + d];
  }
}

// ---------------------------------------------------------------------------
extern "C" void kernel_launch(void* const* d_in, const int* in_sizes, int n_in,
                              void* d_out, int out_size, void* d_ws, size_t ws_size,
                              hipStream_t stream) {
  const float* x      = (const float*)d_in[0];
  const float* qkv_w  = (const float*)d_in[1];
  const float* proj_w = (const float*)d_in[2];
  const float* qg     = (const float*)d_in[3];
  const float* qb     = (const float*)d_in[4];
  const float* kg     = (const float*)d_in[5];
  const float* kb     = (const float*)d_in[6];
  float* out = (float*)d_out;

  char* ws = (char*)d_ws;
  const size_t perbuf = (size_t)16 * NH * NSEQ * HD * sizeof(_Float16);  // 25.2 MB
  const int nx = 16 * NSEQ * CIN;          // 12.58M
  const int nqw = 3 * NH * HD * CIN;       // 1.77M
  const int npw = CIN * CIN;               // 0.59M
  size_t off = 0;
  _Float16* qbuf = (_Float16*)(ws + off); off += perbuf;
  _Float16* kbuf = (_Float16*)(ws + off); off += perbuf;
  _Float16* vtb  = (_Float16*)(ws + off); off += perbuf;
  _Float16* att  = (_Float16*)(ws + off); off += perbuf;
  _Float16* xh   = (_Float16*)(ws + off); off += (size_t)nx * 2;
  _Float16* qwh  = (_Float16*)(ws + off); off += (size_t)nqw * 2;
  _Float16* pwh  = (_Float16*)(ws + off); off += (size_t)npw * 2;

  cvt_f16_kernel<<<nx / 2048, 256, 0, stream>>>(x, xh, nx);
  cvt_f16_kernel<<<nqw / 2048, 256, 0, stream>>>(qkv_w, qwh, nqw);
  cvt_f16_kernel<<<npw / 2048, 256, 0, stream>>>(proj_w, pwh, npw);

  qkv_rope_ln_kernel<<<dim3(36, 128), 256, 0, stream>>>(xh, qwh, qg, qb, kg, kb,
                                                        qbuf, kbuf, vtb);
  flash_attn_kernel<<<dim3(8, 192), 256, 0, stream>>>(qbuf, kbuf, vtb, att);
  proj_kernel<<<dim3(12, 128), 256, 0, stream>>>(att, pwh, out);
}
